// ResFeature_90683939487979
// MI455X (gfx1250) — matrix operand order repaired and checked by reading, not compiled, not run on hardware
//
#include <hip/hip_runtime.h>
#include <hip/hip_bf16.h>
#include <math.h>

typedef __attribute__((ext_vector_type(16))) _Float16 v16h;
typedef __attribute__((ext_vector_type(8)))  _Float16 v8h;
typedef __attribute__((ext_vector_type(8)))  float    v8f;

#define BB 8
#define LL 1024
#define AA 6
#define HH 128
#define KK 32
#define NAa 5
#define SEPS 1e-6f
#define LEPS 1000000.0f

__device__ __forceinline__ float geluf(float x) {
    return 0.5f * x * (1.0f + erff(x * 0.70710678118654752f));
}
__device__ __forceinline__ float clipc(float c) {
    return fminf(fmaxf(c, -1.0f + 1e-6f), 1.0f - 1e-6f);
}

struct F3 { float x, y, z; };
__device__ __forceinline__ F3 f3sub(F3 a, F3 b) { return {a.x-b.x, a.y-b.y, a.z-b.z}; }
__device__ __forceinline__ float f3dot(F3 a, F3 b) { return a.x*b.x + a.y*b.y + a.z*b.z; }
__device__ __forceinline__ F3 f3cross(F3 a, F3 b) {
    return {a.y*b.z - a.z*b.y, a.z*b.x - a.x*b.z, a.x*b.y - a.y*b.x};
}
__device__ __forceinline__ float f3norm(F3 a) { return sqrtf(f3dot(a, a)); }

// Build a 16-half A/B fragment from two aligned 8-half LDS chunks.
__device__ __forceinline__ v16h frag_from(const _Float16* lo, const _Float16* hi) {
    v8h a = *(const v8h*)lo;
    v8h b = *(const v8h*)hi;
    return __builtin_shufflevector(a, b, 0,1,2,3,4,5,6,7,8,9,10,11,12,13,14,15);
}

// ---------------------------------------------------------------------------
// Kernel 1: per-residue raw features, atom-pool MLP, residue MLP -> h (ws)
// Also emits per-residue centers for KNN. 1 block (128 thr) per residue.
// ---------------------------------------------------------------------------
__global__ void residue_kernel(const float* __restrict__ coords,
                               const float* __restrict__ atom_embedding,
                               const float* __restrict__ mask,
                               const float* __restrict__ ap_w0, const float* __restrict__ ap_b0,
                               const float* __restrict__ ap_w1, const float* __restrict__ ap_b1,
                               const float* __restrict__ res_w0, const float* __restrict__ res_b0,
                               const float* __restrict__ res_w1, const float* __restrict__ res_b1,
                               float* __restrict__ ws_h,
                               float* __restrict__ ws_centers) {
    const int resid = blockIdx.x;           // b*L + l
    const int h = threadIdx.x;              // channel 0..127
    const float mval = mask[resid];

    __shared__ float sT[HH];                // MLP hidden
    __shared__ float sP[HH];                // pooled
    __shared__ float sWv[8];                // atom logits

    const float* cp = coords + (size_t)resid * AA * 3;
    F3 c[AA];
#pragma unroll
    for (int a = 0; a < AA; ++a) c[a] = {cp[a*3+0], cp[a*3+1], cp[a*3+2]};

    // raw[10]: 5 dists, 3 angles, 2 dihedrals
    float raw[10];
    F3 u[AA-1];
#pragma unroll
    for (int i = 0; i < AA-1; ++i) {
        u[i] = f3sub(c[i+1], c[i]);
        raw[i] = sqrtf(f3dot(u[i], u[i]) + SEPS);
    }
#pragma unroll
    for (int i = 0; i < NAa-2; ++i) {
        F3 av = {-u[i].x, -u[i].y, -u[i].z};
        F3 bv = u[i+1];
        float cs = f3dot(av, bv) / (f3norm(av) * f3norm(bv) + SEPS);
        raw[5+i] = acosf(clipc(cs));
    }
#pragma unroll
    for (int i = 0; i < NAa-3; ++i) {
        F3 n1 = f3cross(u[i], u[i+1]);
        F3 n2 = f3cross(u[i+1], u[i+2]);
        float inv = 1.0f / (f3norm(u[i+1]) + SEPS);
        F3 u1n = {u[i+1].x*inv, u[i+1].y*inv, u[i+1].z*inv};
        float xx = f3dot(f3cross(n1, n2), u1n);
        float yy = f3dot(n1, n2) + SEPS;
        raw[8+i] = atan2f(xx, yy);
    }

    if (h < 3) {                            // residue center component h
        float s = 0.f;
#pragma unroll
        for (int a = 0; a < AA; ++a) s += cp[a*3 + h];
        ws_centers[(size_t)resid*3 + h] = s / (float)AA;
    }

    // atom-pool MLP: t = gelu(raw @ ap_w0 + b0)
    float acc = ap_b0[h];
#pragma unroll
    for (int r = 0; r < 10; ++r) acc += raw[r] * ap_w0[r*HH + h];
    sT[h] = geluf(acc);
    __syncthreads();

    if (h < AA) {                           // logits w[a] = t @ ap_w1 + b1, masked
        float wa = ap_b1[h];
        for (int cc = 0; cc < HH; ++cc) wa += sT[cc] * ap_w1[cc*AA + h];
        sWv[h] = wa * mval;
    }
    __syncthreads();

    // softmax over atoms (each thread locally)
    float wl[AA], wmax = -3.0e38f;
#pragma unroll
    for (int a = 0; a < AA; ++a) { wl[a] = sWv[a]; wmax = fmaxf(wmax, wl[a]); }
    float wsum = 0.f;
#pragma unroll
    for (int a = 0; a < AA; ++a) { wl[a] = expf(wl[a] - wmax); wsum += wl[a]; }
    float winv = 1.0f / wsum;

    // pooled embedding
    const float* ae = atom_embedding + (size_t)resid * AA * HH;
    float pl = 0.f;
#pragma unroll
    for (int a = 0; a < AA; ++a) pl += ae[a*HH + h] * (wl[a] * winv);
    sP[h] = pl;
    __syncthreads();

    // residue MLP layer 1: gelu([raw, pooled] @ res_w0 + b0)
    acc = res_b0[h];
#pragma unroll
    for (int r = 0; r < 10; ++r) acc += raw[r] * res_w0[r*HH + h];
    for (int cc = 0; cc < HH; ++cc) acc += sP[cc] * res_w0[(10+cc)*HH + h];
    float t2 = geluf(acc);
    __syncthreads();
    sT[h] = t2;
    __syncthreads();

    // layer 2
    acc = res_b1[h];
    for (int cc = 0; cc < HH; ++cc) acc += sT[cc] * res_w1[cc*HH + h];
    ws_h[(size_t)resid*HH + h] = geluf(acc);
}

// ---------------------------------------------------------------------------
// Kernel 2: masked per-(batch, channel) mean / rstd over residues
// ---------------------------------------------------------------------------
__global__ void stats_kernel(const float* __restrict__ ws_h,
                             const float* __restrict__ mask,
                             float* __restrict__ ws_stats) {
    const int b = blockIdx.x, ch = threadIdx.x;
    float s = 0.f, ss = 0.f, cnt = 0.f;
    for (int l = 0; l < LL; ++l) {
        float mm = mask[b*LL + l];
        float v = ws_h[((size_t)b*LL + l)*HH + ch] * mm;
        s += v; ss += v*v; cnt += mm;
    }
    float c = (cnt == 0.f) ? 1.f : cnt;
    float mean = s / c;
    float var = (ss - 2.f*mean*s + (float)LL*mean*mean) / c;
    ws_stats[(b*HH + ch)*2 + 0] = mean;
    ws_stats[(b*HH + ch)*2 + 1] = 1.0f / sqrtf(var + SEPS);
}

// ---------------------------------------------------------------------------
// Kernel 3: apply group-norm -> res_emb output
// ---------------------------------------------------------------------------
__global__ void norm_kernel(const float* __restrict__ ws_h,
                            const float* __restrict__ mask,
                            const float* __restrict__ ws_stats,
                            const float* __restrict__ gn_scale,
                            const float* __restrict__ gn_shift,
                            float* __restrict__ out_res) {
    const int resid = blockIdx.x, ch = threadIdx.x;
    const int b = resid / LL;
    float hv = ws_h[(size_t)resid*HH + ch];
    float mean = ws_stats[(b*HH + ch)*2 + 0];
    float rstd = ws_stats[(b*HH + ch)*2 + 1];
    out_res[(size_t)resid*HH + ch] =
        ((hv - mean) * rstd * gn_scale[ch] + gn_shift[ch]) * mask[resid];
}

// ---------------------------------------------------------------------------
// Kernel 4: KNN (K=32) over residue centers. One wave per residue.
// ---------------------------------------------------------------------------
__global__ void knn_kernel(const float* __restrict__ ws_centers,
                           const float* __restrict__ mask,
                           int* __restrict__ ei_out) {
    const int wv = threadIdx.x >> 5;
    const int lane = threadIdx.x & 31;
    const int resid = blockIdx.x * 8 + wv;
    const int b = resid / LL;
    const int l = resid % LL;

    __shared__ float sDist[8][LL];

    const float cx = ws_centers[(size_t)resid*3+0];
    const float cy = ws_centers[(size_t)resid*3+1];
    const float cz = ws_centers[(size_t)resid*3+2];
    const float ml = mask[resid];

    for (int j = lane; j < LL; j += 32) {
        const float* cj = ws_centers + ((size_t)b*LL + j)*3;
        float dx = cx - cj[0], dy = cy - cj[1], dz = cz - cj[2];
        float d = sqrtf(dx*dx + dy*dy + dz*dz + SEPS);
        float m2 = ml * mask[b*LL + j];
        float v = d*m2 + (1.0f - m2)*LEPS;
        if (j == l) v += LEPS;
        sDist[wv][j] = v;
    }
    __syncthreads();

    for (int kk = 0; kk < KK; ++kk) {
        float bv = 3.0e38f; int bi = 0;
#pragma unroll
        for (int i = 0; i < 32; ++i) {
            int j = i*32 + lane;
            float v = sDist[wv][j];
            if (v < bv) { bv = v; bi = j; }       // keeps smallest index on tie
        }
#pragma unroll
        for (int off = 16; off > 0; off >>= 1) {  // wave32 butterfly
            float ov = __shfl_xor(bv, off, 32);
            int   oi = __shfl_xor(bi, off, 32);
            if (ov < bv || (ov == bv && oi < bi)) { bv = ov; bi = oi; }
        }
        if (lane == 0) {
            ei_out[(size_t)resid*KK + kk] = bi;
            sDist[wv][bi] = 3.0e38f;
        }
        __syncthreads();
    }
}

// ---------------------------------------------------------------------------
// Kernel 5: edge raw features (61) + 2-layer MLP via v_wmma_f32_16x16x32_f16.
// Block = 128 threads (4 waves) = 2 residues * 32 edges = 64 GEMM rows.
// Each wave owns one 16-row tile across 8 column tiles of N=128.
// Weights staged TRANSPOSED in LDS so every B fragment is one aligned
// 32-byte contiguous v16h load (2x ds_load_b128) instead of 16x ds_load_u16.
// LDS: union{ A(8KB)+W0t(16KB) | W1t(32KB) } + mid(16KB) + biases ~= 49.5KB.
// ---------------------------------------------------------------------------
__global__ void edge_kernel(const float* __restrict__ coords,
                            const int* __restrict__ ei,
                            const float* __restrict__ edge_w0, const float* __restrict__ edge_b0,
                            const float* __restrict__ edge_w1, const float* __restrict__ edge_b1,
                            float* __restrict__ out_edge) {
    const int tid  = threadIdx.x;
    const int wv   = tid >> 5;
    const int lane = tid & 31;
    const int m    = lane & 15;        // row-in-tile (A) / column (B,C)
    const int g    = lane >> 4;        // lane-group

    const int resid0 = blockIdx.x * 2; // two residues per block (same batch)

    __shared__ __attribute__((aligned(32))) _Float16 sU[16384];   // 32KB union
    __shared__ __attribute__((aligned(32))) _Float16 sMid[64*HH]; // 16KB hidden (f16)
    __shared__ float sB0[HH], sB1[HH];

    _Float16* sA   = sU;               // 64 rows x 64 cols (row-major)
    _Float16* sW0t = sU + 64*64;       // transposed: [n=0..127][k=0..63]
    _Float16* sW1t = sU;               // phase 2, transposed: [n=0..127][k=0..127]

    __builtin_prefetch(edge_w1, 0, 0); // warm GL2 for phase-2 weights

    // ---- stage W0 transposed (f32 -> f16, zero-pad K rows 61..63) + biases ----
    for (int idx = tid; idx < 64*HH; idx += 128) {
        int kk = idx >> 7, n = idx & 127;          // coalesced global read
        sW0t[n*64 + kk] = (kk < 61) ? (_Float16)edge_w0[kk*HH + n] : (_Float16)0.0f;
    }
    sB0[tid] = edge_b0[tid];
    sB1[tid] = edge_b1[tid];

    // ---- compute 61 raw edge features, one thread per edge row ----
    if (tid < 64) {
        const int r      = tid;
        const int resid  = resid0 + (r >> 5);
        const int b      = resid / LL;
        const int k      = r & 31;
        const int jj     = ei[(size_t)resid*KK + k];
        const int residj = b*LL + jj;

        const float* cpi = coords + (size_t)resid  * AA * 3;
        const float* cpj = coords + (size_t)residj * AA * 3;
        F3 ci[AA], cj[AA];
#pragma unroll
        for (int a = 0; a < AA; ++a) {
            ci[a] = {cpi[a*3], cpi[a*3+1], cpi[a*3+2]};
            cj[a] = {cpj[a*3], cpj[a*3+1], cpj[a*3+2]};
        }
        _Float16* row = sA + r*64;
        // 36 cross-distances, feature p*6+q
#pragma unroll
        for (int p = 0; p < AA; ++p)
#pragma unroll
            for (int q = 0; q < AA; ++q) {
                F3 d = f3sub(ci[p], cj[q]);
                row[p*AA + q] = (_Float16)sqrtf(f3dot(d, d) + SEPS);
            }
        F3 ui[4], uj[4];
        float ni[4], nj[4];
#pragma unroll
        for (int p = 0; p < 4; ++p) {
            ui[p] = f3sub(ci[p+1], ci[p]); ni[p] = f3norm(ui[p]);
            uj[p] = f3sub(cj[p+1], cj[p]); nj[p] = f3norm(uj[p]);
        }
        // 16 cross-angles, feature 36 + p*4+q
#pragma unroll
        for (int p = 0; p < 4; ++p)
#pragma unroll
            for (int q = 0; q < 4; ++q) {
                float cs = f3dot(ui[p], uj[q]) / (ni[p]*nj[q] + SEPS);
                row[36 + p*4 + q] = (_Float16)acosf(clipc(cs));
            }
        F3 mi[3], mj[3];
        float nmi[3], nmj[3];
#pragma unroll
        for (int p = 0; p < 3; ++p) {
            mi[p] = f3cross(ui[p], ui[p+1]); nmi[p] = f3norm(mi[p]);
            mj[p] = f3cross(uj[p], uj[p+1]); nmj[p] = f3norm(mj[p]);
        }
        // 9 cross-dihedral angles, feature 52 + p*3+q
#pragma unroll
        for (int p = 0; p < 3; ++p)
#pragma unroll
            for (int q = 0; q < 3; ++q) {
                float cs = f3dot(mi[p], mj[q]) / (nmi[p]*nmj[q] + SEPS);
                row[52 + p*3 + q] = (_Float16)acosf(clipc(cs));
            }
        row[61] = (_Float16)0.0f; row[62] = (_Float16)0.0f; row[63] = (_Float16)0.0f;
    }
    __syncthreads();

    // ---- GEMM1: (64x64) @ (64x128), K in 2 steps of 32, fused bias+GELU ----
    {
        const int rbase = wv * 16;
        const _Float16* arow = sA + (rbase + m)*64;
        v16h a0 = frag_from(arow +      g*16, arow +  8 + g*16);   // ks=0
        v16h a1 = frag_from(arow + 32 + g*16, arow + 40 + g*16);   // ks=1
#pragma unroll
        for (int ct = 0; ct < 8; ++ct) {
            const int col = ct*16 + m;
            const _Float16* bcol = sW0t + col*64;
            v8f acc = {};
            v16h bf0 = *(const v16h*)(bcol +      g*16);         // K = g*16 + e
            acc = __builtin_amdgcn_wmma_f32_16x16x32_f16(false, a0, false, bf0,
                                                         (short)0, acc, false, false);
            v16h bf1 = *(const v16h*)(bcol + 32 + g*16);         // K = 32 + g*16 + e
            acc = __builtin_amdgcn_wmma_f32_16x16x32_f16(false, a1, false, bf1,
                                                         (short)0, acc, false, false);
            const float bias = sB0[col];
#pragma unroll
            for (int v = 0; v < 8; ++v) {
                int rrow = rbase + g*8 + v;
                sMid[rrow*HH + col] = (_Float16)geluf(acc[v] + bias);
            }
        }
    }
    __syncthreads();

    // ---- stage W1 transposed over the union region ----
    for (int idx = tid; idx < HH*HH; idx += 128) {
        int kk = idx >> 7, n = idx & 127;          // coalesced global read
        sW1t[n*HH + kk] = (_Float16)edge_w1[kk*HH + n];
    }
    __syncthreads();

    // ---- GEMM2: (64x128) @ (128x128), K in 4 steps, bias+GELU, store f32 ----
    {
        const int rbase = wv * 16;
        const _Float16* arow = sMid + (rbase + m)*HH;
        v16h A2[4];
#pragma unroll
        for (int ks = 0; ks < 4; ++ks)
            A2[ks] = frag_from(arow + ks*32 + g*16, arow + ks*32 + 8 + g*16);
#pragma unroll
        for (int ct = 0; ct < 8; ++ct) {
            const int col = ct*16 + m;
            const _Float16* bcol = sW1t + col*HH;
            v8f acc = {};
#pragma unroll
            for (int ks = 0; ks < 4; ++ks) {
                v16h bf = *(const v16h*)(bcol + ks*32 + g*16);
                acc = __builtin_amdgcn_wmma_f32_16x16x32_f16(false, A2[ks], false, bf,
                                                             (short)0, acc, false, false);
            }
            const float bias = sB1[col];
#pragma unroll
            for (int v = 0; v < 8; ++v) {
                int rrow = rbase + g*8 + v;                // row within block
                size_t edge = (size_t)resid0*KK + rrow;    // global edge row
                out_edge[edge*HH + col] = geluf(acc[v] + bias);
            }
        }
    }
}

// ---------------------------------------------------------------------------
extern "C" void kernel_launch(void* const* d_in, const int* in_sizes, int n_in,
                              void* d_out, int out_size, void* d_ws, size_t ws_size,
                              hipStream_t stream) {
    const float* atom_embedding = (const float*)d_in[0];
    // d_in[1] = atom_mask (== mask repeated per atom; mask used directly)
    const float* coords   = (const float*)d_in[2];
    const float* mask     = (const float*)d_in[3];
    const float* ap_w0    = (const float*)d_in[4];
    const float* ap_b0    = (const float*)d_in[5];
    const float* ap_w1    = (const float*)d_in[6];
    const float* ap_b1    = (const float*)d_in[7];
    const float* res_w0   = (const float*)d_in[8];
    const float* res_b0   = (const float*)d_in[9];
    const float* res_w1   = (const float*)d_in[10];
    const float* res_b1   = (const float*)d_in[11];
    const float* gn_scale = (const float*)d_in[12];
    const float* gn_shift = (const float*)d_in[13];
    const float* edge_w0  = (const float*)d_in[14];
    const float* edge_b0  = (const float*)d_in[15];
    const float* edge_w1  = (const float*)d_in[16];
    const float* edge_b1  = (const float*)d_in[17];

    const size_t RES_N  = (size_t)BB*LL*HH;        // 1,048,576
    const size_t EDGE_N = (size_t)BB*LL*KK*HH;     // 33,554,432

    float* out_res  = (float*)d_out;
    float* out_edge = out_res + RES_N;
    int*   out_ei   = (int*)(out_res + RES_N + EDGE_N);

    float* ws         = (float*)d_ws;
    float* ws_h       = ws;                         // B*L*H
    float* ws_stats   = ws_h + RES_N;               // B*H*2
    float* ws_centers = ws_stats + (size_t)BB*HH*2; // B*L*3

    residue_kernel<<<BB*LL, HH, 0, stream>>>(coords, atom_embedding, mask,
                                             ap_w0, ap_b0, ap_w1, ap_b1,
                                             res_w0, res_b0, res_w1, res_b1,
                                             ws_h, ws_centers);
    stats_kernel<<<BB, HH, 0, stream>>>(ws_h, mask, ws_stats);
    norm_kernel<<<BB*LL, HH, 0, stream>>>(ws_h, mask, ws_stats,
                                          gn_scale, gn_shift, out_res);
    knn_kernel<<<(BB*LL)/8, 256, 0, stream>>>(ws_centers, mask, out_ei);
    edge_kernel<<<(BB*LL)/2, 128, 0, stream>>>(coords, out_ei,
                                               edge_w0, edge_b0, edge_w1, edge_b1,
                                               out_edge);
}
